// Model_36670430773656
// MI455X (gfx1250) — matrix operand order repaired and checked by reading, not compiled, not run on hardware
//
#include <hip/hip_runtime.h>
#include <stddef.h>

// ---- problem constants ----
#define VOCAB 30000
#define EMB   100
#define HID   128
#define T_SEQ 200
#define BATCH 512
#define NCLS  9
#define FOURH 512          // 4*HID
#define KTOT  256          // padded-x (128) + h (128)
#define MTILE 16
#define BN_EPS 1e-3f

typedef __attribute__((ext_vector_type(16))) __bf16 v16bf;
typedef __attribute__((ext_vector_type(8)))  __bf16 v8bf;
typedef __attribute__((ext_vector_type(8)))  float  v8f;

__device__ __forceinline__ float sigmoidf_(float x) {
    return 1.0f / (1.0f + __expf(-x));
}

// ---------------------------------------------------------------------------
// 1) Embedding gather -> bf16, padded to 128 (slot 100 = 1.0 carries bias)
// ---------------------------------------------------------------------------
__global__ void embed_kernel(const int* __restrict__ tokens,
                             const float* __restrict__ emb,
                             __bf16* __restrict__ xpad) {
    const int bt = blockIdx.x;          // 0..B*T-1
    const int k  = threadIdx.x;         // 0..127
    const int tok = tokens[bt];
    float v;
    if (k < EMB)       v = emb[(size_t)tok * EMB + k];
    else if (k == EMB) v = 1.0f;        // bias slot
    else               v = 0.0f;
    xpad[(size_t)bt * 128 + k] = (__bf16)v;
}

// ---------------------------------------------------------------------------
// 2) Build combined [256 x 512] weight = [W ; bias ; 0 ; U], pre-swizzled to
//    the WMMA bf16 B-operand lane layout:
//    dst[((ntile*8 + kb)*32 + lane)*16 + e] = Wc[kb*32 + (lane<16?0:8) + (e<8 ? e : 8 + e)][ntile*16 + lane%16]
// ---------------------------------------------------------------------------
__global__ void swizzle_w_kernel(const float* __restrict__ Wf, const float* __restrict__ Uf,
                                 const float* __restrict__ bf, const float* __restrict__ Wb,
                                 const float* __restrict__ Ub, const float* __restrict__ bb,
                                 __bf16* __restrict__ wcf, __bf16* __restrict__ wcb) {
    const int idx = blockIdx.x * blockDim.x + threadIdx.x;   // 0..131071
    const int e     = idx & 15;
    const int lane  = (idx >> 4) & 31;
    const int kb    = (idx >> 9) & 7;
    const int ntile = idx >> 12;                             // 0..31
    const int n = ntile * 16 + (lane & 15);
    const int k = kb * 32 + ((lane < 16) ? 0 : 8) + ((e < 8) ? e : 8 + e);
    float vf, vb;
    if (k < EMB)        { vf = Wf[k * FOURH + n];         vb = Wb[k * FOURH + n]; }
    else if (k == EMB)  { vf = bf[n];                     vb = bb[n]; }
    else if (k < HID)   { vf = 0.0f;                      vb = 0.0f; }
    else                { vf = Uf[(k - HID) * FOURH + n]; vb = Ub[(k - HID) * FOURH + n]; }
    wcf[idx] = (__bf16)vf;
    wcb[idx] = (__bf16)vb;
}

// ---------------------------------------------------------------------------
// 3) Fold BatchNorm (inference) into the dense layer
// ---------------------------------------------------------------------------
__global__ void bnfold_kernel(const float* __restrict__ gamma, const float* __restrict__ beta,
                              const float* __restrict__ mean,  const float* __restrict__ var,
                              const float* __restrict__ Wd,    const float* __restrict__ bd,
                              float* __restrict__ wd_eff, float* __restrict__ bd_eff) {
    const int tid = threadIdx.x;   // 256 threads, one per hidden channel
    const float sc = gamma[tid] * rsqrtf(var[tid] + BN_EPS);
    for (int c = 0; c < NCLS; ++c)
        wd_eff[tid * NCLS + c] = Wd[tid * NCLS + c] * sc;
    if (tid < NCLS) {
        float s = bd[tid];
        for (int k = 0; k < 2 * HID; ++k) {
            float sck = gamma[k] * rsqrtf(var[k] + BN_EPS);
            s += (beta[k] - mean[k] * sck) * Wd[k * NCLS + tid];
        }
        bd_eff[tid] = s;
    }
}

// ---------------------------------------------------------------------------
// 4) LSTM scan. One workgroup = 16 batch rows x one direction, full T loop.
//    256 threads = 8 waves. Wave w owns ntiles {w, 8+w, 16+w, 24+w}, i.e. the
//    SAME 16 columns of each gate block (i,f,g,o) -> gates + c-state stay in
//    registers. Weights VGPR-resident for the whole scan (256 VGPRs/wave).
//    Ping-pong h buffer in LDS -> single barrier per step. All per-step
//    addresses are pointer-stepped (constant stride), no index recompute.
// ---------------------------------------------------------------------------
__global__ __launch_bounds__(256)
void lstm_scan_kernel(const __bf16* __restrict__ xpad,
                      const __bf16* __restrict__ wcf,
                      const __bf16* __restrict__ wcb,
                      __bf16* __restrict__ hstore) {
    __shared__ __bf16 h_lds[2][MTILE * HID];   // 8 KB ping-pong h state

    const int tid   = threadIdx.x;
    const int lane  = tid & 31;
    const int wv    = tid >> 5;
    const int dir   = blockIdx.y;            // 0 = fwd, 1 = bwd
    const int bbase = blockIdx.x * MTILE;

    // A/B/D lane geometry
    const int m      = lane & 15;            // A: row; B/D: column within tile
    const int hiHalf = lane >> 4;
    const int ka0    = hiHalf ? 8  : 0;      // A 16-bit 16x32 layout chunk offsets
    const int ka1    = hiHalf ? 24 : 16;
    const int Mo     = hiHalf ? 8 : 0;       // D: element r -> row Mo + r
    const int j      = wv * 16 + m;          // gate-local column owned by this lane

    // ---- load this wave's B fragments into registers (resident for all T) ----
    // breg[g][kb]: gate g (i,f,g,o) -> ntile g*8 + wv
    const __bf16* wsrc = dir ? wcb : wcf;
    v16bf breg[4][8];
    #pragma unroll
    for (int g = 0; g < 4; ++g) {
        #pragma unroll
        for (int kb = 0; kb < 8; ++kb) {
            const int ntile = g * 8 + wv;
            breg[g][kb] = *(const v16bf*)(wsrc + (((ntile * 8 + kb) * 32 + lane) << 4));
        }
    }

    // zero initial h state (buffer 0)
    for (int i = tid; i < MTILE * HID; i += 256) h_lds[0][i] = (__bf16)0.0f;

    // c-state lives in registers: element r <-> (row Mo+r, column j)
    v8f cst = {};

    // ---- pointer-stepped addresses (constant stride per step) ----
    const int t0 = dir ? (T_SEQ - 1) : 0;
    const ptrdiff_t xstride = dir ? -(ptrdiff_t)128 : (ptrdiff_t)128;
    const ptrdiff_t hstride = (dir ? -(ptrdiff_t)1 : (ptrdiff_t)1) * HID * BATCH;

    const __bf16* xptr = xpad + ((size_t)(bbase + m) * T_SEQ + t0) * 128;
    __bf16*       hptr = hstore + (((size_t)dir * T_SEQ + t0) * HID + j) * BATCH + bbase + Mo;

    // preload x fragments for the first step
    v8bf xreg[8];
    #pragma unroll
    for (int kb = 0; kb < 4; ++kb) {
        xreg[2 * kb + 0] = *(const v8bf*)(xptr + kb * 32 + ka0);
        xreg[2 * kb + 1] = *(const v8bf*)(xptr + kb * 32 + ka1);
    }
    __syncthreads();

    #pragma unroll 1
    for (int step = 0; step < T_SEQ; ++step) {
        const __bf16* hrd = h_lds[step & 1];
        __bf16*       hwr = h_lds[(step & 1) ^ 1];

        // ---- build A fragments: kb 0..3 from x (registers), 4..7 from h (LDS)
        union { v16bf v; v8bf h[2]; } afrag[8];
        #pragma unroll
        for (int kb = 0; kb < 4; ++kb) {
            afrag[kb].h[0] = xreg[2 * kb + 0];
            afrag[kb].h[1] = xreg[2 * kb + 1];
        }
        const __bf16* hrow = hrd + m * HID;
        #pragma unroll
        for (int kb = 0; kb < 4; ++kb) {
            afrag[4 + kb].h[0] = *(const v8bf*)(hrow + kb * 32 + ka0);
            afrag[4 + kb].h[1] = *(const v8bf*)(hrow + kb * 32 + ka1);
        }

        // ---- prefetch next step's x into xreg (overlaps with WMMA)
        if (step + 1 < T_SEQ) xptr += xstride;   // last step: re-reads same t (harmless)
        #pragma unroll
        for (int kb = 0; kb < 4; ++kb) {
            xreg[2 * kb + 0] = *(const v8bf*)(xptr + kb * 32 + ka0);
            xreg[2 * kb + 1] = *(const v8bf*)(xptr + kb * 32 + ka1);
        }

        // ---- WMMA: acc[g] accumulates gate g's 16x16 tile; kb outer for ILP
        v8f acc[4] = {{}, {}, {}, {}};
        #pragma unroll
        for (int kb = 0; kb < 8; ++kb) {
            #pragma unroll
            for (int g = 0; g < 4; ++g) {
                acc[g] = __builtin_amdgcn_wmma_f32_16x16x32_bf16(
                            false, afrag[kb].v, false, breg[g][kb], (short)0, acc[g], false, false);
            }
        }

        // ---- gates entirely in registers: element r is (row Mo+r, column j)
        v8bf hvec;
        #pragma unroll
        for (int r = 0; r < 8; ++r) {
            const float ig = sigmoidf_(acc[0][r]);
            const float fg = sigmoidf_(acc[1][r]);
            const float g  = fmaxf(acc[2][r], 0.0f);   // activation='relu' on candidate
            const float og = sigmoidf_(acc[3][r]);
            const float c  = fg * cst[r] + ig * g;
            cst[r] = c;
            const float h = og * fmaxf(c, 0.0f);       // activation='relu' on cell out
            hvec[r] = (__bf16)h;
        }

        // ---- write h: ping-pong LDS (next step's A) + global [dir][t][j][b]
        #pragma unroll
        for (int r = 0; r < 8; ++r) hwr[(Mo + r) * HID + j] = hvec[r];
        *(v8bf*)hptr = hvec;
        hptr += hstride;

        // single barrier: writes to buffer p^1 visible before next step reads it
        __syncthreads();
    }
}

// ---------------------------------------------------------------------------
// 5) Epilogue: (BN-folded) dense + softmax. One thread per (b,t); coalesced
//    reads of hstore[dir][t][k][b].
// ---------------------------------------------------------------------------
__global__ void head_kernel(const __bf16* __restrict__ hstore,
                            const float* __restrict__ wd_eff,
                            const float* __restrict__ bd_eff,
                            float* __restrict__ out) {
    __shared__ float wsh[2 * HID * NCLS];
    __shared__ float bsh[NCLS];
    const int tid = threadIdx.x;
    for (int i = tid; i < 2 * HID * NCLS; i += 256) wsh[i] = wd_eff[i];
    if (tid < NCLS) bsh[tid] = bd_eff[tid];
    __syncthreads();

    const int idx = blockIdx.x * 256 + tid;   // 0..B*T-1
    const int b = idx & (BATCH - 1);          // consecutive threads -> consecutive b
    const int t = idx >> 9;                   // BATCH == 512

    float acc[NCLS];
    #pragma unroll
    for (int c = 0; c < NCLS; ++c) acc[c] = bsh[c];

    const __bf16* hf = hstore + ((size_t)t * HID) * BATCH + b;
    const __bf16* hb = hstore + (((size_t)T_SEQ + t) * HID) * BATCH + b;
    for (int k = 0; k < HID; ++k) {
        const float v = (float)hf[(size_t)k * BATCH];
        #pragma unroll
        for (int c = 0; c < NCLS; ++c) acc[c] = fmaf(v, wsh[k * NCLS + c], acc[c]);
    }
    for (int k = 0; k < HID; ++k) {
        const float v = (float)hb[(size_t)k * BATCH];
        #pragma unroll
        for (int c = 0; c < NCLS; ++c) acc[c] = fmaf(v, wsh[(HID + k) * NCLS + c], acc[c]);
    }

    float mx = acc[0];
    #pragma unroll
    for (int c = 1; c < NCLS; ++c) mx = fmaxf(mx, acc[c]);
    float s = 0.0f;
    #pragma unroll
    for (int c = 0; c < NCLS; ++c) { acc[c] = __expf(acc[c] - mx); s += acc[c]; }
    const float inv = 1.0f / s;
    float* op = out + ((size_t)b * T_SEQ + t) * NCLS;
    #pragma unroll
    for (int c = 0; c < NCLS; ++c) op[c] = acc[c] * inv;
}

// ---------------------------------------------------------------------------
// launch
// ---------------------------------------------------------------------------
extern "C" void kernel_launch(void* const* d_in, const int* in_sizes, int n_in,
                              void* d_out, int out_size, void* d_ws, size_t ws_size,
                              hipStream_t stream) {
    const int*   tokens = (const int*)  d_in[0];
    const float* emb    = (const float*)d_in[1];
    const float* Wf     = (const float*)d_in[2];
    const float* Uf     = (const float*)d_in[3];
    const float* bf     = (const float*)d_in[4];
    const float* Wb     = (const float*)d_in[5];
    const float* Ub     = (const float*)d_in[6];
    const float* bb     = (const float*)d_in[7];
    const float* gamma  = (const float*)d_in[8];
    const float* beta   = (const float*)d_in[9];
    const float* mmean  = (const float*)d_in[10];
    const float* mvar   = (const float*)d_in[11];
    const float* Wd     = (const float*)d_in[12];
    const float* bd     = (const float*)d_in[13];

    // workspace layout (bytes)
    char* ws = (char*)d_ws;
    const size_t off_x  = 0;                                   // B*T*128 bf16 = 26,214,400
    const size_t off_wf = off_x  + (size_t)BATCH * T_SEQ * 128 * 2;
    const size_t off_wb = off_wf + (size_t)KTOT * FOURH * 2;   // 262,144 each
    const size_t off_h  = off_wb + (size_t)KTOT * FOURH * 2;   // 2*T*HID*B bf16 = 52,428,800
    const size_t off_wd = off_h  + (size_t)2 * T_SEQ * HID * BATCH * 2;
    const size_t off_bd = off_wd + (size_t)2 * HID * NCLS * 4;

    __bf16* xpad   = (__bf16*)(ws + off_x);
    __bf16* wcf    = (__bf16*)(ws + off_wf);
    __bf16* wcb    = (__bf16*)(ws + off_wb);
    __bf16* hstore = (__bf16*)(ws + off_h);
    float*  wd_eff = (float*) (ws + off_wd);
    float*  bd_eff = (float*) (ws + off_bd);

    embed_kernel<<<BATCH * T_SEQ, 128, 0, stream>>>(tokens, emb, xpad);
    swizzle_w_kernel<<<(KTOT * FOURH) / 256, 256, 0, stream>>>(Wf, Uf, bf, Wb, Ub, bb, wcf, wcb);
    bnfold_kernel<<<1, 256, 0, stream>>>(gamma, beta, mmean, mvar, Wd, bd, wd_eff, bd_eff);
    lstm_scan_kernel<<<dim3(BATCH / MTILE, 2), 256, 0, stream>>>(xpad, wcf, wcb, hstore);
    head_kernel<<<(BATCH * T_SEQ) / 256, 256, 0, stream>>>(hstore, wd_eff, bd_eff, (float*)d_out);
}
